// Multihead_self_attention_block_15255723836045
// MI455X (gfx1250) — compile-verified, matrix operand order checked by reading, not run on hardware
//
#include <hip/hip_runtime.h>
#include <hip/hip_bf16.h>
#include <math.h>

// ---------- problem constants ----------
#define BS_    4
#define NQ_    4096
#define DM_    512
#define HEADS_ 8
#define DK_    64
#define NK_    1024           // after stride-2 reduction: 32*32
#define H_IMG_ 64
#define EPS_   1e-5f

typedef __attribute__((ext_vector_type(16))) __bf16 v16bf;
typedef __attribute__((ext_vector_type(8)))  float  v8f;

// ---------- CDNA5 async global->LDS path (guarded; falls back to sync) ----
#if defined(__HIP_DEVICE_COMPILE__) && \
    __has_builtin(__builtin_amdgcn_global_load_async_to_lds_b128) && \
    __has_builtin(__builtin_amdgcn_s_wait_asynccnt)
#define HAS_ASYNC_LDS 1
#else
#define HAS_ASYNC_LDS 0
#endif

#if HAS_ASYNC_LDS
typedef __attribute__((__vector_size__(4 * sizeof(int)))) int v4i_t;
typedef __attribute__((address_space(1))) v4i_t as1_v4i;
typedef __attribute__((address_space(3))) v4i_t as3_v4i;
#endif

__device__ __forceinline__ void cp16_async(const __bf16* g, __bf16* l) {
#if HAS_ASYNC_LDS
  __builtin_amdgcn_global_load_async_to_lds_b128(
      (as1_v4i*)g, (as3_v4i*)l, 0, 0);
#else
  *(uint4*)l = *(const uint4*)g;
#endif
}
__device__ __forceinline__ void cp_wait() {
#if HAS_ASYNC_LDS
  __builtin_amdgcn_s_wait_asynccnt(0);
#endif
}

// ---------- WMMA helper (device-guarded) ----------
__device__ __forceinline__ v8f wmma_bf16(v16bf a, v16bf b, v8f c) {
#if defined(__HIP_DEVICE_COMPILE__)
  return __builtin_amdgcn_wmma_f32_16x16x32_bf16(false, a, false, b, (short)0, c, false, false);
#else
  return c;
#endif
}

// Load a 16x32 A-frag (or B-frag from a transposed tile) from LDS.
// p must already point at lds[row][col0 + klo] where klo = (lane&16)?8:0.
__device__ __forceinline__ v16bf ld_frag(const __bf16* p) {
  union { v16bf v; uint4 u[2]; } r;
  r.u[0] = *(const uint4*)p;
  r.u[1] = *(const uint4*)(p + 16);
  return r.v;
}

// =====================================================================
// k0: fp32 -> bf16 convert
// =====================================================================
__global__ __launch_bounds__(256) void f2bf_kernel(const float* __restrict__ in,
                                                   __bf16* __restrict__ out, int n) {
  int i = blockIdx.x * 256 + threadIdx.x;
  if (i < n) out[i] = (__bf16)in[i];
}

// =====================================================================
// k1: bf16 GEMM  C(Mx512) = A(Mx512) * B(512x512) + bias
//     WMMA 16x16x32 bf16, block tile 128x128, 8 waves (2x4), BK=32,
//     double-buffered LDS with async global->LDS for A.
// =====================================================================
#define GBM 128
#define GBN 128
#define GBK 32
#define GST (GBK + 8)    // lds row stride (bf16): 80B -> 16B-aligned frags
#define GKI 16           // 512 / 32 k-iterations

template <int OUTF32>
__global__ __launch_bounds__(256) void gemm512_kernel(
    const __bf16* __restrict__ A, const __bf16* __restrict__ B,
    const float* __restrict__ bias, void* __restrict__ Cout, int M) {
  __shared__ __bf16 As[2][GBM][GST];
  __shared__ __bf16 Bts[2][GBN][GST];   // B transposed: Bts[buf][n][k]

  const int tid  = threadIdx.x;
  const int wave = tid >> 5;
  const int lane = tid & 31;
  const int wm   = wave >> 2;           // 0..1
  const int wn   = wave & 3;            // 0..3
  const int bm   = blockIdx.x * GBM;
  const int bn   = blockIdx.y * GBN;
  const int klo  = (lane & 16) ? 8 : 0;
  const int ln15 = lane & 15;
  const int hl   = lane >> 4;

  const int arow = tid >> 1, ach = (tid & 1) * 16;
  const int bkr  = tid >> 3, bcs = (tid & 7) * 16;
  const __bf16* Ag = A + ((bm + arow) << 9) + ach;   // advance by k0
  const __bf16* Bg = B + (bkr << 9) + bn + bcs;      // advance by k0*512

  v8f acc[4][2];
#pragma unroll
  for (int mf = 0; mf < 4; ++mf)
#pragma unroll
    for (int nf = 0; nf < 2; ++nf) acc[mf][nf] = {};

  // ---- prologue: tile 0 ----
  cp16_async(Ag,     &As[0][arow][ach]);
  cp16_async(Ag + 8, &As[0][arow][ach + 8]);
  {
    uint4 b0 = *(const uint4*)Bg;
    uint4 b1 = *(const uint4*)(Bg + 8);
    __bf16 e[16];
    *(uint4*)&e[0] = b0; *(uint4*)&e[8] = b1;
#pragma unroll
    for (int i = 0; i < 16; ++i) Bts[0][bcs + i][bkr] = e[i];
  }
  cp_wait();
  __syncthreads();

  // ---- pipelined main loop ----
  for (int it = 0; it < GKI; ++it) {
    const int cur = it & 1, nxt = cur ^ 1;
    const bool haveNext = (it + 1) < GKI;
    uint4 b0r{}, b1r{};
    if (haveNext) {
      const __bf16* An = Ag + ((it + 1) << 5);
      cp16_async(An,     &As[nxt][arow][ach]);
      cp16_async(An + 8, &As[nxt][arow][ach + 8]);
      const __bf16* Bn = Bg + ((it + 1) << 14);      // +32 rows of 512
      b0r = *(const uint4*)Bn;
      b1r = *(const uint4*)(Bn + 8);
    }

    v16bf bfr[2];
#pragma unroll
    for (int nf = 0; nf < 2; ++nf)
      bfr[nf] = ld_frag(&Bts[cur][wn * 32 + nf * 16 + ln15][klo]);
#pragma unroll
    for (int mf = 0; mf < 4; ++mf) {
      v16bf afr = ld_frag(&As[cur][wm * 64 + mf * 16 + ln15][klo]);
#pragma unroll
      for (int nf = 0; nf < 2; ++nf)
        acc[mf][nf] = wmma_bf16(afr, bfr[nf], acc[mf][nf]);
    }

    if (haveNext) {
      __bf16 e[16];
      *(uint4*)&e[0] = b0r; *(uint4*)&e[8] = b1r;
#pragma unroll
      for (int i = 0; i < 16; ++i) Bts[nxt][bcs + i][bkr] = e[i];
      cp_wait();
    }
    __syncthreads();
  }

  // ---- branchless epilogue ----
#pragma unroll
  for (int nf = 0; nf < 2; ++nf) {
    const int col = bn + wn * 32 + nf * 16 + ln15;
    const float bv = bias[col];
#pragma unroll
    for (int mf = 0; mf < 4; ++mf) {
      const int row0 = bm + wm * 64 + mf * 16 + 8 * hl;
#pragma unroll
      for (int r = 0; r < 8; ++r) {
        const int off = ((row0 + r) << 9) + col;
        const float v = acc[mf][nf][r] + bv;
        if (OUTF32) ((float*)Cout)[off]  = v;
        else        ((__bf16*)Cout)[off] = (__bf16)v;
      }
    }
  }
}

// =====================================================================
// k2: depthwise 3x3 stride-2 conv + bias + LayerNorm(512) -> bf16 x
// =====================================================================
__global__ __launch_bounds__(256) void dwconv_ln_kernel(
    const float* __restrict__ q, const float* __restrict__ cw,
    const float* __restrict__ cb, const float* __restrict__ lw,
    const float* __restrict__ lb, __bf16* __restrict__ out) {
  const int b   = blockIdx.y;
  const int pos = blockIdx.x;          // 0..1023
  const int oy  = pos >> 5, ox = pos & 31;
  const int tid = threadIdx.x;
  __shared__ float red[256];

  float v[2];
#pragma unroll
  for (int j = 0; j < 2; ++j) {
    int c = tid + j * 256;
    float acc = cb[c];
#pragma unroll
    for (int ky = 0; ky < 3; ++ky) {
      int iy = oy * 2 - 1 + ky;
      if (iy < 0 || iy >= H_IMG_) continue;
#pragma unroll
      for (int kx = 0; kx < 3; ++kx) {
        int ix = ox * 2 - 1 + kx;
        if (ix < 0 || ix >= H_IMG_) continue;
        acc += cw[c * 9 + ky * 3 + kx] *
               q[(((b << 12) + (iy << 6) + ix) << 9) + c];
      }
    }
    v[j] = acc;
  }
  red[tid] = v[0] + v[1];
  __syncthreads();
  for (int st = 128; st > 0; st >>= 1) {
    if (tid < st) red[tid] += red[tid + st];
    __syncthreads();
  }
  float mu = red[0] * (1.f / 512.f);
  __syncthreads();
  float d0 = v[0] - mu, d1 = v[1] - mu;
  red[tid] = d0 * d0 + d1 * d1;
  __syncthreads();
  for (int st = 128; st > 0; st >>= 1) {
    if (tid < st) red[tid] += red[tid + st];
    __syncthreads();
  }
  float rs = rsqrtf(red[0] * (1.f / 512.f) + EPS_);
  int base = ((b << 10) + pos) << 9;
  out[base + tid]       = (__bf16)(d0 * rs * lw[tid]       + lb[tid]);
  out[base + tid + 256] = (__bf16)(d1 * rs * lw[tid + 256] + lb[tid + 256]);
}

// =====================================================================
// k3: column-sum of V (per b, per channel) + zero attsq accumulator
// =====================================================================
__global__ __launch_bounds__(256) void colsum_v_kernel(const __bf16* __restrict__ V,
                                                       float* __restrict__ colsum,
                                                       float* __restrict__ attsq) {
  int b = blockIdx.y;
  int c = blockIdx.x * 256 + threadIdx.x;   // grid.x = 2
  float s = 0.f;
  for (int k = 0; k < NK_; ++k) s += (float)V[(((b << 10) + k) << 9) + c];
  colsum[(b << 9) + c] = s;
  if (blockIdx.x == 0 && blockIdx.y == 0 && threadIdx.x < BS_ * HEADS_)
    attsq[threadIdx.x] = 0.f;
}

// =====================================================================
// k4: fused flash-attention: head-mix + online softmax + Σatt² + PV
//     block = 16 q rows x all 8 heads (wave w = head w); KT=32 k-tile
// =====================================================================
#define QT 16
#define KT 32
#define QSST (DM_ + 8)   // 520
#define VTST (KT + 8)    // 40
#define PSST (KT + 8)    // 40

__global__ __launch_bounds__(256) void attn_kernel(
    const __bf16* __restrict__ Qp, const __bf16* __restrict__ Kp,
    const __bf16* __restrict__ Vp, const float* __restrict__ tw,
    const float* __restrict__ tb, float* __restrict__ Obuf,
    float* __restrict__ attsq) {
  __shared__ __bf16 Qs[QT][QSST];
  __shared__ __bf16 Ks[KT][QSST];
  __shared__ __bf16 Vts[DM_][VTST];          // V transposed: Vts[d][kk]
  __shared__ float  raws[HEADS_][QT][KT + 1];
  __shared__ __bf16 Ps[HEADS_][QT][PSST];

  const int tid  = threadIdx.x;
  const int wave = tid >> 5;
  const int lane = tid & 31;
  const int o    = wave;                     // head handled by this wave
  const int b    = blockIdx.y;
  const int q0   = blockIdx.x * QT;
  const int klo  = (lane & 16) ? 8 : 0;
  const int ln15 = lane & 15;
  const int hl   = lane >> 4;

  // Q tile: 16 rows x 512, async to LDS
  {
    int row = tid >> 4, cs = (tid & 15) * 32;
    const __bf16* gp = Qp + (((b << 12) + q0 + row) << 9) + cs;
#pragma unroll
    for (int g = 0; g < 4; ++g)
      cp16_async(gp + g * 8, &Qs[row][cs + g * 8]);
  }
  cp_wait();
  __syncthreads();

  float wq[HEADS_];
#pragma unroll
  for (int h = 0; h < HEADS_; ++h) wq[h] = tw[o * HEADS_ + h] * 0.125f;
  const float obias = tb[o];

  // hoisted Q A-frags for this head (d-chunks 0..31, 32..63)
  v16bf aq0 = ld_frag(&Qs[ln15][o * DK_ + klo]);
  v16bf aq1 = ld_frag(&Qs[ln15][o * DK_ + 32 + klo]);

  float mrun[8], lrun[8], l2run[8];
#pragma unroll
  for (int r = 0; r < 8; ++r) { mrun[r] = -1e30f; lrun[r] = 0.f; l2run[r] = 0.f; }
  v8f oacc[4];
#pragma unroll
  for (int nf = 0; nf < 4; ++nf) oacc[nf] = {};

  const int vrow = tid >> 3, vcs = (tid & 7) * 64;  // tile-load lanes
  for (int kt = 0; kt < NK_ / KT; ++kt) {
    const int k0 = kt * KT;
    { // K tile 32x512: async to LDS
      const __bf16* gp = Kp + (((b << 10) + k0 + vrow) << 9) + vcs;
#pragma unroll
      for (int g = 0; g < 8; ++g)
        cp16_async(gp + g * 8, &Ks[vrow][vcs + g * 8]);
    }
    { // V tile 32x512 -> transposed LDS (through regs)
      const __bf16* gp = Vp + (((b << 10) + k0 + vrow) << 9) + vcs;
      __builtin_prefetch(gp + KT * DM_, 0, 0);
#pragma unroll
      for (int g = 0; g < 8; ++g) {
        uint4 u = *(const uint4*)(gp + g * 8);
        __bf16 e[8];
        *(uint4*)e = u;
#pragma unroll
        for (int i = 0; i < 8; ++i) Vts[vcs + g * 8 + i][vrow] = e[i];
      }
    }
    cp_wait();
    __syncthreads();   // S1

    // raw scores for input head h = wave: raw = Q_h (16x64) @ K_h^T (64x32)
#pragma unroll
    for (int nf = 0; nf < 2; ++nf) {
      v16bf bk0 = ld_frag(&Ks[nf * 16 + ln15][o * DK_ + klo]);
      v16bf bk1 = ld_frag(&Ks[nf * 16 + ln15][o * DK_ + 32 + klo]);
      v8f c = {};
      c = wmma_bf16(aq0, bk0, c);
      c = wmma_bf16(aq1, bk1, c);
#pragma unroll
      for (int r = 0; r < 8; ++r)
        raws[o][r + 8 * hl][nf * 16 + ln15] = c[r];
    }
    __syncthreads();   // S2

    // head-mix into registers: mixed[o] = sum_h w[o,h]*raw[h]/8 + b[o]
    float mx[2][8];
#pragma unroll
    for (int nf = 0; nf < 2; ++nf)
#pragma unroll
      for (int r = 0; r < 8; ++r) mx[nf][r] = obias;
#pragma unroll
    for (int h = 0; h < HEADS_; ++h)
#pragma unroll
      for (int nf = 0; nf < 2; ++nf)
#pragma unroll
        for (int r = 0; r < 8; ++r)
          mx[nf][r] += wq[h] * raws[h][r + 8 * hl][nf * 16 + ln15];
    __syncthreads();   // S3 (raws free again)

    // online softmax over this 32-column slab
#pragma unroll
    for (int r = 0; r < 8; ++r) {
      float s0 = mx[0][r], s1 = mx[1][r];
      float pm = fmaxf(s0, s1);
#pragma unroll
      for (int mk = 1; mk < 16; mk <<= 1) pm = fmaxf(pm, __shfl_xor(pm, mk, 32));
      float nm = fmaxf(mrun[r], pm);
      float al = __expf(mrun[r] - nm);
      float p0 = __expf(s0 - nm), p1 = __expf(s1 - nm);
      float ls = p0 + p1, l2s = p0 * p0 + p1 * p1;
#pragma unroll
      for (int mk = 1; mk < 16; mk <<= 1) {
        ls  += __shfl_xor(ls, mk, 32);
        l2s += __shfl_xor(l2s, mk, 32);
      }
      lrun[r]  = lrun[r] * al + ls;
      l2run[r] = l2run[r] * (al * al) + l2s;
      mrun[r]  = nm;
#pragma unroll
      for (int nf = 0; nf < 4; ++nf) oacc[nf][r] *= al;
      Ps[o][r + 8 * hl][ln15]      = (__bf16)p0;
      Ps[o][r + 8 * hl][16 + ln15] = (__bf16)p1;
    }

    // PV: O += P (16x32) @ V_o (32x64)
    v16bf ap = ld_frag(&Ps[o][ln15][klo]);
#pragma unroll
    for (int nf = 0; nf < 4; ++nf) {
      v16bf bv = ld_frag(&Vts[o * DK_ + nf * 16 + ln15][klo]);
      oacc[nf] = wmma_bf16(ap, bv, oacc[nf]);
    }
    __syncthreads();   // S4 (Ks/Vts free for next iter)
  }

  // finalize: O/l, store fp32; accumulate sum att^2 per (b,head)
#pragma unroll
  for (int r = 0; r < 8; ++r) {
    float inv = 1.f / lrun[r];
    const int row = (b << 12) + q0 + r + 8 * hl;
#pragma unroll
    for (int nf = 0; nf < 4; ++nf)
      Obuf[(row << 9) + o * DK_ + nf * 16 + ln15] = oacc[nf][r] * inv;
  }
  float s2 = 0.f;
#pragma unroll
  for (int r = 0; r < 8; ++r) s2 += l2run[r] / (lrun[r] * lrun[r]);
  if (ln15 == 0) atomicAdd(&attsq[b * HEADS_ + o], s2);
}

// =====================================================================
// k5: InstanceNorm correction -> bf16 for the output projection
// =====================================================================
__global__ __launch_bounds__(256) void scale_o_kernel(
    const float* __restrict__ O, const float* __restrict__ colsum,
    const float* __restrict__ attsq, __bf16* __restrict__ out) {
  int i = blockIdx.x * 256 + threadIdx.x;
  int c = i & (DM_ - 1);
  int b = i >> 21;                          // / (4096*512)
  int h = c >> 6;
  const float inv_nk = 1.f / (float)NK_;
  float m2 = attsq[b * HEADS_ + h] * (1.f / ((float)NQ_ * (float)NK_));
  float vr = m2 - inv_nk * inv_nk;
  float val = (O[i] - inv_nk * colsum[(b << 9) + c]) * rsqrtf(vr + EPS_);
  out[i] = (__bf16)val;
}

// =====================================================================
// host launcher
// =====================================================================
extern "C" void kernel_launch(void* const* d_in, const int* in_sizes, int n_in,
                              void* d_out, int out_size, void* d_ws, size_t ws_size,
                              hipStream_t stream) {
  (void)in_sizes; (void)n_in; (void)out_size; (void)ws_size;
  const float* queries = (const float*)d_in[0];
  const float* Wq = (const float*)d_in[3];
  const float* bq = (const float*)d_in[4];
  const float* Wk = (const float*)d_in[5];
  const float* bk = (const float*)d_in[6];
  const float* Wv = (const float*)d_in[7];
  const float* bv = (const float*)d_in[8];
  const float* Wo = (const float*)d_in[9];
  const float* bo = (const float*)d_in[10];
  const float* conv_w = (const float*)d_in[11];
  const float* conv_b = (const float*)d_in[12];
  const float* ln_w = (const float*)d_in[13];
  const float* ln_b = (const float*)d_in[14];
  const float* tconv_w = (const float*)d_in[15];
  const float* tconv_b = (const float*)d_in[16];

  char* p = (char*)d_ws;
  auto alloc = [&](size_t bytes) {
    void* r = (void*)p;
    p += (bytes + 255) & ~(size_t)255;
    return r;
  };
  const size_t nQ = (size_t)BS_ * NQ_ * DM_;     // 8,388,608
  const size_t nX = (size_t)BS_ * NK_ * DM_;     // 2,097,152
  const size_t nW = (size_t)DM_ * DM_;           // 262,144

  __bf16* qbf  = (__bf16*)alloc(nQ * 2);
  __bf16* wqb  = (__bf16*)alloc(nW * 2);
  __bf16* wkb  = (__bf16*)alloc(nW * 2);
  __bf16* wvb  = (__bf16*)alloc(nW * 2);
  __bf16* wob  = (__bf16*)alloc(nW * 2);
  __bf16* Qp   = (__bf16*)alloc(nQ * 2);
  __bf16* xbf  = (__bf16*)alloc(nX * 2);
  __bf16* Kp   = (__bf16*)alloc(nX * 2);
  __bf16* Vp   = (__bf16*)alloc(nX * 2);
  float*  csum = (float*) alloc((size_t)BS_ * DM_ * 4);
  float*  asq  = (float*) alloc((size_t)BS_ * HEADS_ * 4);
  float*  Obuf = (float*) alloc(nQ * 4);
  __bf16* Obf  = (__bf16*)alloc(nQ * 2);

  // 0) converts
  f2bf_kernel<<<dim3((int)((nQ + 255) / 256)), 256, 0, stream>>>(queries, qbf, (int)nQ);
  f2bf_kernel<<<dim3((int)((nW + 255) / 256)), 256, 0, stream>>>(Wq, wqb, (int)nW);
  f2bf_kernel<<<dim3((int)((nW + 255) / 256)), 256, 0, stream>>>(Wk, wkb, (int)nW);
  f2bf_kernel<<<dim3((int)((nW + 255) / 256)), 256, 0, stream>>>(Wv, wvb, (int)nW);
  f2bf_kernel<<<dim3((int)((nW + 255) / 256)), 256, 0, stream>>>(Wo, wob, (int)nW);

  // 1) Q projection (16384x512x512) -> bf16
  gemm512_kernel<0><<<dim3((BS_ * NQ_) / GBM, DM_ / GBN), 256, 0, stream>>>(
      qbf, wqb, bq, Qp, BS_ * NQ_);

  // 2) depthwise conv + LN -> x (4,1024,512) bf16
  dwconv_ln_kernel<<<dim3(NK_, BS_), 256, 0, stream>>>(queries, conv_w, conv_b,
                                                       ln_w, ln_b, xbf);

  // 3) K and V projections (4096x512x512) -> bf16
  gemm512_kernel<0><<<dim3((BS_ * NK_) / GBM, DM_ / GBN), 256, 0, stream>>>(
      xbf, wkb, bk, Kp, BS_ * NK_);
  gemm512_kernel<0><<<dim3((BS_ * NK_) / GBM, DM_ / GBN), 256, 0, stream>>>(
      xbf, wvb, bv, Vp, BS_ * NK_);

  // 4) colsum(V) + zero attsq
  colsum_v_kernel<<<dim3(DM_ / 256, BS_), 256, 0, stream>>>(Vp, csum, asq);

  // 5) fused attention (head-mix + online softmax + sum att^2 + PV)
  attn_kernel<<<dim3(NQ_ / QT, BS_), 256, 0, stream>>>(Qp, Kp, Vp, tconv_w,
                                                       tconv_b, Obuf, asq);

  // 6) InstanceNorm correction -> bf16
  scale_o_kernel<<<dim3((int)((nQ + 255) / 256)), 256, 0, stream>>>(Obuf, csum, asq, Obf);

  // 7) output projection -> fp32 d_out
  gemm512_kernel<1><<<dim3((BS_ * NQ_) / GBM, DM_ / GBN), 256, 0, stream>>>(
      Obf, wob, bo, (float*)d_out, BS_ * NQ_);
}
